// RCSLS_31911607009474
// MI455X (gfx1250) — matrix operand order repaired and checked by reading, not compiled, not run on hardware
//
#include <hip/hip_runtime.h>
#include <hip/hip_bf16.h>

// Problem constants (from reference setup_inputs)
#define DIM    512
#define NROWS  4096
#define NCOLS  32768
#define MROWS  32      // rows per block (two 16-row WMMA tiles per wave)
#define AS     520     // A LDS row stride in bf16 elements (conflict-free b128)
#define STG    18      // staging row stride in floats (conflict-free b64 scans)
#define NWAVES 8

typedef __attribute__((ext_vector_type(16))) __bf16 v16bf;
typedef __attribute__((ext_vector_type(8)))  __bf16 v8bf;
typedef __attribute__((ext_vector_type(8)))  float  v8f;

// Pack two f32 -> packed bf16x2 in one u32.
__device__ __forceinline__ unsigned pack2(float lo, float hi) {
#if __has_builtin(__builtin_amdgcn_cvt_pk_bf16_f32)
    auto r = __builtin_amdgcn_cvt_pk_bf16_f32(lo, hi);   // v_cvt_pk_bf16_f32
    return __builtin_bit_cast(unsigned, r);
#else
    union { float f; unsigned u; } a, b; a.f = lo; b.f = hi;
    // round-half-up, then pick the two high halves with one v_perm_b32
    return __builtin_amdgcn_perm(b.u + 0x8000u, a.u + 0x8000u, 0x07060302u);
#endif
}

// t[] sorted descending; precondition s > t[9]. Branchless insertion network.
__device__ __forceinline__ void topk_insert(float (&t)[10], float s) {
#pragma unroll
    for (int i = 0; i < 10; ++i) {
        float hi = fmaxf(t[i], s);
        s        = fminf(t[i], s);
        t[i] = hi;
    }
}

// One-shot f32 -> bf16 conversion of a Z matrix into workspace.
__global__ __launch_bounds__(256) void rcsls_cvt(
    const float* __restrict__ src, uint2* __restrict__ dst, int n4)
{
    const float4* s4 = (const float4*)src;
    for (int i = blockIdx.x * blockDim.x + threadIdx.x; i < n4;
         i += gridDim.x * blockDim.x) {
        float4 v = s4[i];
        uint2 o;
        o.x = pack2(v.x, v.y);
        o.y = pack2(v.z, v.w);
        dst[i] = o;
    }
}

// Fused  scores = A(32 rows) x Z^T  +  per-row running top-10, streamed over N.
// PREB=true : Z is pre-converted bf16 (row-major [NCOLS][DIM]).
// PREB=false: Z is f32, converted on the fly.
template <bool PREB>
__global__ __launch_bounds__(256) void rcsls_topk_gemm(
    const float* __restrict__ A0, const void* __restrict__ Z0,
    const float* __restrict__ A1, const void* __restrict__ Z1,
    float* __restrict__ acc)
{
    __shared__ __align__(16) __bf16 Alds[MROWS * AS];            // 33.3 KB
    __shared__ __align__(16) float  stage[NWAVES * MROWS * STG]; // 18.4 KB
    __shared__ float mergebuf[NWAVES * MROWS * 10];              // 10.2 KB
    __shared__ float red[MROWS];

    const int job = blockIdx.x >> 7;
    const int rb  = blockIdx.x & 127;
    const float*  __restrict__ A  = job ? A1 : A0;
    const void*   __restrict__ Zv = job ? Z1 : Z0;
    const __bf16* __restrict__ ZB = (const __bf16*)Zv;
    const float*  __restrict__ ZF = (const float*)Zv;

    const int tid  = threadIdx.x;
    const int lane = tid & 31;
    const int wid  = tid >> 5;
    const int m    = lane & 15;   // row-in-tile (A) / col-in-tile (B)
    const int hi   = lane >> 4;   // lane-group select

    // Stage this block's 32 A rows into LDS as bf16 (reused 2048 times).
    {
        const float* Ab = A + (size_t)rb * MROWS * DIM;
        for (int i = tid; i < MROWS * DIM / 2; i += 256) {
            int idx = i << 1;
            int r = idx >> 9, k = idx & 511;
            const float2 v = *(const float2*)(Ab + r * DIM + k);
            *(unsigned*)&Alds[r * AS + k] = pack2(v.x, v.y);
        }
    }
    __syncthreads();

    float t[10];
#pragma unroll
    for (int i = 0; i < 10; ++i) t[i] = -1e30f;

    float* stg = stage + wid * MROWS * STG;

    for (int tile = wid; tile < NCOLS / 16; tile += NWAVES) {
        const int n = tile * 16 + m;

        v8f c0 = {}, c1 = {};
#pragma unroll 4
        for (int ks = 0; ks < DIM / 32; ++ks) {
            const int kb = ks * 32;

            // B fragment (shared by both M-tiles): lane holds column n,
            // contiguous K in [hi*16, +16).
            v16bf b;
            if constexpr (PREB) {
                const __bf16* pz = ZB + (size_t)n * DIM + kb + hi * 16;
                v8bf b0 = *(const v8bf*)(pz);
                v8bf b1 = *(const v8bf*)(pz + 8);
#pragma unroll
                for (int i = 0; i < 8; ++i) { b[i] = b0[i]; b[8 + i] = b1[i]; }
            } else {
                const float* pz = ZF + (size_t)n * DIM + kb + hi * 16;
                float4 z0 = *(const float4*)(pz + 0);
                float4 z1 = *(const float4*)(pz + 4);
                float4 z2 = *(const float4*)(pz + 8);
                float4 z3 = *(const float4*)(pz + 12);
                union { unsigned u[8]; v16bf v; } bb;
                bb.u[0] = pack2(z0.x, z0.y); bb.u[1] = pack2(z0.z, z0.w);
                bb.u[2] = pack2(z1.x, z1.y); bb.u[3] = pack2(z1.z, z1.w);
                bb.u[4] = pack2(z2.x, z2.y); bb.u[5] = pack2(z2.z, z2.w);
                bb.u[6] = pack2(z3.x, z3.y); bb.u[7] = pack2(z3.z, z3.w);
                b = bb.v;
            }

            // A fragments: contiguous K in [hi*16, +16) (matches B)
            const __bf16* pa0 = Alds + m * AS + kb + hi * 16;         // rows 0..15
            const __bf16* pa1 = Alds + (m + 16) * AS + kb + hi * 16;  // rows 16..31
            v8bf a0l = *(const v8bf*)(pa0);
            v8bf a0h = *(const v8bf*)(pa0 + 8);
            v8bf a1l = *(const v8bf*)(pa1);
            v8bf a1h = *(const v8bf*)(pa1 + 8);
            v16bf a0, a1;
#pragma unroll
            for (int i = 0; i < 8; ++i) {
                a0[i] = a0l[i]; a0[8 + i] = a0h[i];
                a1[i] = a1l[i]; a1[8 + i] = a1h[i];
            }

            c0 = __builtin_amdgcn_wmma_f32_16x16x32_bf16(
                     false, a0, false, b, (short)0, c0, false, false);
            c1 = __builtin_amdgcn_wmma_f32_16x16x32_bf16(
                     false, a1, false, b, (short)0, c1, false, false);
        }

        // Prefetch this lane's next-tile row (near-cache locality).
        if (tile + NWAVES < NCOLS / 16) {
            if constexpr (PREB)
                __builtin_prefetch(ZB + (size_t)(n + NWAVES * 16) * DIM, 0, 3);
            else
                __builtin_prefetch(ZF + (size_t)(n + NWAVES * 16) * DIM, 0, 3);
        }

        // Spill both C tiles to LDS: VGPR v holds row (v + hi*8), col m.
#pragma unroll
        for (int v = 0; v < 8; ++v) {
            stg[(v + hi * 8) * STG + m]      = c0[v];
            stg[(16 + v + hi * 8) * STG + m] = c1[v];
        }
        __syncthreads();

        // All 32 lanes: scan own row (lane = row 0..31), update running top-10.
        {
            const float2* srow = (const float2*)(stg + lane * STG);
#pragma unroll
            for (int q = 0; q < 8; ++q) {
                float2 s2 = srow[q];
                if (s2.x > t[9]) topk_insert(t, s2.x);
                if (s2.y > t[9]) topk_insert(t, s2.y);
            }
        }
        __syncthreads();
    }

    // Merge the 8 per-wave top-10 lists per row, sum, reduce, atomic-accumulate.
    {
        float* mb = mergebuf + (wid * MROWS + lane) * 10;
#pragma unroll
        for (int i = 0; i < 10; ++i) mb[i] = t[i];
    }
    __syncthreads();
    if (tid < MROWS) {
        float t2[10];
#pragma unroll
        for (int i = 0; i < 10; ++i) t2[i] = -1e30f;
        for (int w = 0; w < NWAVES; ++w) {
            const float* mb = mergebuf + (w * MROWS + tid) * 10;
#pragma unroll
            for (int i = 0; i < 10; ++i) {
                float s = mb[i];
                if (s > t2[9]) topk_insert(t2, s);
            }
        }
        float rs = 0.f;
#pragma unroll
        for (int i = 0; i < 10; ++i) rs += t2[i];
        red[tid] = rs;
    }
    __syncthreads();
    if (tid == 0) {
        float s = 0.f;
#pragma unroll
        for (int i = 0; i < MROWS; ++i) s += red[i];
        atomicAdd(acc, s);   // global_atomic_add_f32
    }
}

__global__ __launch_bounds__(256) void rcsls_dot(
    const float* __restrict__ a, const float* __restrict__ b,
    int n4, float* __restrict__ acc)
{
    __shared__ float red[256];
    const float4* a4 = (const float4*)a;
    const float4* b4 = (const float4*)b;
    float s = 0.f;
    for (int i = blockIdx.x * blockDim.x + threadIdx.x; i < n4;
         i += gridDim.x * blockDim.x) {
        float4 x = a4[i], y = b4[i];
        s += x.x * y.x + x.y * y.y + x.z * y.z + x.w * y.w;
    }
    red[threadIdx.x] = s;
    __syncthreads();
    for (int off = 128; off > 0; off >>= 1) {
        if (threadIdx.x < off) red[threadIdx.x] += red[threadIdx.x + off];
        __syncthreads();
    }
    if (threadIdx.x == 0) atomicAdd(acc, red[0]);
}

__global__ void rcsls_finalize(const float* __restrict__ ws,
                               const int* __restrict__ knn,
                               float* __restrict__ out)
{
    float fk  = ws[0] / (float)knn[0];   // fk0 + fk1
    float dot = ws[1];
    out[0] = (fk - 2.0f * dot) / (float)NROWS;   // -(2*dot - fk0 - fk1)/B
}

extern "C" void kernel_launch(void* const* d_in, const int* in_sizes, int n_in,
                              void* d_out, int out_size, void* d_ws, size_t ws_size,
                              hipStream_t stream) {
    // setup_inputs order: X_src, X_trans, Y_tgt, Z_src, Z_trans, Z_tgt, knn
    const float* X_trans = (const float*)d_in[1];
    const float* Y_tgt   = (const float*)d_in[2];
    const float* Z_trans = (const float*)d_in[4];
    const float* Z_tgt   = (const float*)d_in[5];
    const int*   knn     = (const int*)d_in[6];

    float* acc = (float*)d_ws;   // acc[0] = sum of top-k values, acc[1] = dot
    hipMemsetAsync(d_ws, 0, 2 * sizeof(float), stream);

    rcsls_dot<<<256, 256, 0, stream>>>(X_trans, Y_tgt, NROWS * DIM / 4, acc + 1);

    const size_t zbytes = (size_t)NCOLS * DIM * sizeof(__bf16);  // 32 MB per matrix
    if (ws_size >= 256 + 2 * zbytes) {
        // Pre-convert both Z matrices to bf16 in workspace (L2-resident, 64 MB).
        uint2* bz0 = (uint2*)((char*)d_ws + 256);
        uint2* bz1 = (uint2*)((char*)d_ws + 256 + zbytes);
        rcsls_cvt<<<2048, 256, 0, stream>>>(Z_tgt,   bz0, NCOLS * DIM / 4);
        rcsls_cvt<<<2048, 256, 0, stream>>>(Z_trans, bz1, NCOLS * DIM / 4);
        rcsls_topk_gemm<true><<<256, 256, 0, stream>>>(
            X_trans, (const void*)bz0, Y_tgt, (const void*)bz1, acc);
    } else {
        // Workspace too small: convert B fragments on the fly.
        rcsls_topk_gemm<false><<<256, 256, 0, stream>>>(
            X_trans, (const void*)Z_tgt, Y_tgt, (const void*)Z_trans, acc);
    }

    rcsls_finalize<<<1, 1, 0, stream>>>(acc, knn, (float*)d_out);
}
